// xLSTMBackbone_86225763435051
// MI455X (gfx1250) — matrix operand order repaired and checked by reading, not compiled, not run on hardware
//
#include <hip/hip_runtime.h>

// ---------------------------------------------------------------------------
// Types for CDNA5 WMMA (wave32)
// ---------------------------------------------------------------------------
typedef __attribute__((ext_vector_type(16))) __bf16 v16bf;
typedef __attribute__((ext_vector_type(8)))  float  v8f;
typedef __attribute__((ext_vector_type(4)))  unsigned int u32x4;
typedef __attribute__((ext_vector_type(4)))  float  f32x4;

union Frag {
  u32x4        q[2];
  unsigned int u[8];
  v16bf        v;
};

__device__ __forceinline__ unsigned short f2bf(float f) {
  unsigned int u = __float_as_uint(f);
  u += 0x7FFFu + ((u >> 16) & 1u);     // round-to-nearest-even
  return (unsigned short)(u >> 16);
}

__device__ __forceinline__ v8f wmma_bf16(v16bf a, v16bf b, v8f c) {
  return __builtin_amdgcn_wmma_f32_16x16x32_bf16(false, a, false, b, (short)0, c,
                                                 false, false);
}

// Async global -> LDS copy of 32 bytes (2 x b128), tracked by ASYNCcnt.
// ldsaddr = low 32 bits of generic pointer to __shared__ (== LDS offset).
__device__ __forceinline__ void async_copy32(unsigned ldsaddr,
                                             const unsigned short* gptr) {
  asm volatile(
      "global_load_async_to_lds_b128 %0, %1, off\n\t"
      "global_load_async_to_lds_b128 %0, %1, off offset:16"
      :: "v"(ldsaddr), "v"(gptr) : "memory");
}

#define NTOK   4096     // B*S
#define SEQ    1024
#define DMODEL 512
#define DINNER 1024
#define NHEAD  4
#define DHM    256      // mLSTM head dim
#define DHS    128      // sLSTM head dim
#define FFDIM  704

// ---------------------------------------------------------------------------
// WMMA GEMM:  C[M x N] = A[M x K](bf16,row) @ Bt[N x K](bf16,row)^T
// M,N multiples of 64; K multiple of 32; lda/ldb multiples of 8.
// Double-buffered LDS fed by GLOBAL_LOAD_ASYNC_TO_LDS_B128 (ASYNCcnt),
// one barrier per K-step. Optional bias[N], residual add, bf16 copy of C.
// ---------------------------------------------------------------------------
__global__ __launch_bounds__(128) void k_gemm(
    const unsigned short* __restrict__ A, int lda,
    const unsigned short* __restrict__ Bt, int ldb,
    float* __restrict__ C, int ldc,
    const float* __restrict__ bias,
    const float* __restrict__ res,
    unsigned short* __restrict__ Cb,
    int K)
{
  __shared__ __align__(16) unsigned short As[2][64][40];
  __shared__ __align__(16) unsigned short Bs[2][64][40];
  const unsigned BUFSTRIDE = 64u * 40u * 2u;   // bytes per buffer (5120)

  const int n0 = blockIdx.x * 64;
  const int m0 = blockIdx.y * 64;
  const int tid = threadIdx.x;
  const int wave = tid >> 5, lane = tid & 31;
  const int half = lane >> 4, ln = lane & 15;
  const int lrow = tid >> 1, lcol = (tid & 1) * 16;   // 128 thr cover 64x32 tile

  const unsigned short* gA = A + (size_t)(m0 + lrow) * lda + lcol;
  const unsigned short* gB = Bt + (size_t)(n0 + lrow) * ldb + lcol;
  const unsigned ldsA0 = (unsigned)(size_t)&As[0][lrow][lcol];
  const unsigned ldsB0 = (unsigned)(size_t)&Bs[0][lrow][lcol];

  v8f acc[4] = {};
  const int nk = K >> 5;

  // prologue: issue async loads for tile 0
  async_copy32(ldsA0, gA);
  async_copy32(ldsB0, gB);

  for (int kt = 0; kt < nk; kt++) {
    asm volatile("s_wait_asynccnt 0" ::: "memory");  // own async stores landed
    __syncthreads();                                  // everyone's landed
    if (kt + 1 < nk) {                                // prefetch next tile
      unsigned buf = (unsigned)((kt + 1) & 1) * BUFSTRIDE;
      async_copy32(ldsA0 + buf, gA + (kt + 1) * 32);
      async_copy32(ldsB0 + buf, gB + (kt + 1) * 32);
    }
    const int buf = kt & 1;
    // load A-frag and all four B-frags into distinct registers first so the
    // backend can keep many ds_load_b128 in flight with partial dscnt waits,
    // overlapping LDS reads with WMMA issue.
    Frag a, b0, b1, b2, b3;
    a.q[0]  = *(const u32x4*)&As[buf][wave * 16 + ln][16 * half];
    a.q[1]  = *(const u32x4*)&As[buf][wave * 16 + ln][16 * half + 8];
    b0.q[0] = *(const u32x4*)&Bs[buf][0 * 16 + ln][16 * half];
    b0.q[1] = *(const u32x4*)&Bs[buf][0 * 16 + ln][16 * half + 8];
    b1.q[0] = *(const u32x4*)&Bs[buf][1 * 16 + ln][16 * half];
    b1.q[1] = *(const u32x4*)&Bs[buf][1 * 16 + ln][16 * half + 8];
    b2.q[0] = *(const u32x4*)&Bs[buf][2 * 16 + ln][16 * half];
    b2.q[1] = *(const u32x4*)&Bs[buf][2 * 16 + ln][16 * half + 8];
    b3.q[0] = *(const u32x4*)&Bs[buf][3 * 16 + ln][16 * half];
    b3.q[1] = *(const u32x4*)&Bs[buf][3 * 16 + ln][16 * half + 8];
    acc[0] = wmma_bf16(a.v, b0.v, acc[0]);
    acc[1] = wmma_bf16(a.v, b1.v, acc[1]);
    acc[2] = wmma_bf16(a.v, b2.v, acc[2]);
    acc[3] = wmma_bf16(a.v, b3.v, acc[3]);
  }

#pragma unroll
  for (int nt = 0; nt < 4; nt++) {
#pragma unroll
    for (int r = 0; r < 8; r++) {
      int row = m0 + wave * 16 + r + 8 * half;
      int col = n0 + nt * 16 + ln;
      float v = acc[nt][r];
      if (bias) v += bias[col];
      if (res)  v += res[(size_t)row * ldc + col];
      C[(size_t)row * ldc + col] = v;
      if (Cb) Cb[(size_t)row * ldc + col] = f2bf(v);
    }
  }
}

// ---------------------------------------------------------------------------
// Fused mLSTM cell (flash-style, stabilized):
//   cw[s,t] = qk[s,t]*dh^-0.5 * exp(u[t]-M[s]) * (t<=s)
//   h[s]    = sum_t cw[s,t] v[t] / (max(|rowsum|, nf[s]) + 1e-6)
// One wave per 16-row s-tile, full dh=256 accumulator per wave.
// ---------------------------------------------------------------------------
__global__ __launch_bounds__(128) void k_attn(
    const unsigned short* __restrict__ qb,   // (b, s, NH*256) bf16
    const unsigned short* __restrict__ kb,
    const unsigned short* __restrict__ vT,   // per bh: (256, 1024) bf16
    const float* __restrict__ uarr,          // (bh, S)
    const float* __restrict__ marr,
    const float* __restrict__ nfarr,
    float* __restrict__ hout)                // (b, s, NH*256)
{
  __shared__ float cbuf[4][16][33];

  const int bh = blockIdx.x;
  const int b = bh >> 2, h = bh & 3;
  const int wave = threadIdx.x >> 5, lane = threadIdx.x & 31;
  const int half = lane >> 4, ln = lane & 15;
  const int s0 = (blockIdx.y * 4 + wave) * 16;

  const unsigned short* qh = qb + (size_t)b * SEQ * DINNER + h * DHM;
  const unsigned short* kh = kb + (size_t)b * SEQ * DINNER + h * DHM;
  const unsigned short* vh = vT + (size_t)bh * DHM * SEQ;
  const float* uu = uarr + bh * SEQ;
  const float* mm = marr + bh * SEQ;
  const float* nn = nfarr + bh * SEQ;

  // preload q A-fragments for all 8 k-steps (dh = 256)
  Frag qa[8];
  {
    const unsigned short* qrow = qh + (size_t)(s0 + ln) * DINNER;
#pragma unroll
    for (int ks = 0; ks < 8; ks++) {
      qa[ks].q[0] = *(const u32x4*)(qrow + ks * 32 + 16 * half);
      qa[ks].q[1] = *(const u32x4*)(qrow + ks * 32 + 16 * half + 8);
    }
  }

  float Ms[8], nf[8], rs[8];
#pragma unroll
  for (int r = 0; r < 8; r++) {
    int s = s0 + r + 8 * half;
    Ms[r] = mm[s]; nf[r] = nn[s]; rs[r] = 0.f;
  }

  v8f acc[16] = {};
  const float scale = 0.0625f;   // 256^-0.5
  const int tmax = s0 + 16;

  for (int t0 = 0; t0 < tmax; t0 += 32) {
    // ---- qk: two 16x16 score tiles -------------------------------------
    v8f cq[2] = {};
#pragma unroll
    for (int tt = 0; tt < 2; tt++) {
      const unsigned short* krow = kh + (size_t)(t0 + tt * 16 + ln) * DINNER;
#pragma unroll
      for (int ks = 0; ks < 8; ks++) {
        Frag kf;
        kf.q[0] = *(const u32x4*)(krow + ks * 32 + 16 * half);
        kf.q[1] = *(const u32x4*)(krow + ks * 32 + 16 * half + 8);
        cq[tt] = wmma_bf16(qa[ks].v, kf.v, cq[tt]);
      }
    }
    // ---- weight + causal mask + rowsum; stage for layout transpose -----
#pragma unroll
    for (int tt = 0; tt < 2; tt++) {
      int t = t0 + tt * 16 + ln;
      float ut = uu[t];
#pragma unroll
      for (int r = 0; r < 8; r++) {
        int s = s0 + r + 8 * half;
        float wv = 0.f;
        if (t <= s) wv = scale * __expf(ut - Ms[r]);
        float cw = cq[tt][r] * wv;
        rs[r] += cw;
        cbuf[wave][r + 8 * half][tt * 16 + ln] = cw;
      }
    }
    asm volatile("s_wait_dscnt 0" ::: "memory");
    // ---- rebuild scores as bf16 A-fragment (16 x 32) -------------------
    Frag ca;
#pragma unroll
    for (int j = 0; j < 8; j++) {
      int k0a = 16 * half + 2 * j;
      unsigned lo = f2bf(cbuf[wave][ln][k0a]);
      unsigned hi = f2bf(cbuf[wave][ln][k0a + 1]);
      ca.u[j] = lo | (hi << 16);
    }
    // ---- cw @ v over dh = 256 (16 n-subtiles) --------------------------
#pragma unroll
    for (int nt = 0; nt < 16; nt++) {
      const unsigned short* vrow = vh + (size_t)(nt * 16 + ln) * SEQ + t0;
      Frag vf;
      vf.q[0] = *(const u32x4*)(vrow + 16 * half);
      vf.q[1] = *(const u32x4*)(vrow + 16 * half + 8);
      acc[nt] = wmma_bf16(ca.v, vf.v, acc[nt]);
    }
  }

  // ---- rowsum reduce across 16 lanes of each half, normalize, store ----
  float norm[8];
#pragma unroll
  for (int r = 0; r < 8; r++) {
    float v = rs[r];
    v += __shfl_xor(v, 1, 32);
    v += __shfl_xor(v, 2, 32);
    v += __shfl_xor(v, 4, 32);
    v += __shfl_xor(v, 8, 32);
    norm[r] = fmaxf(fabsf(v), nf[r]) + 1e-6f;
  }
#pragma unroll
  for (int nt = 0; nt < 16; nt++) {
#pragma unroll
    for (int r = 0; r < 8; r++) {
      int s = s0 + r + 8 * half;
      hout[((size_t)(b * SEQ + s)) * DINNER + h * DHM + nt * 16 + ln] =
          acc[nt][r] / norm[r];
    }
  }
}

// ---------------------------------------------------------------------------
// LayerNorm / multi-head groupnorm: per token, groups of size G, scale w[D],
// optional residual add, optional bf16 copy. One wave per token.
// ---------------------------------------------------------------------------
__global__ __launch_bounds__(256) void k_ln(
    const float* __restrict__ x, const float* __restrict__ w,
    float* __restrict__ out, unsigned short* __restrict__ outb,
    const float* __restrict__ res, int T, int D, int G)
{
  int wv = threadIdx.x >> 5, lane = threadIdx.x & 31;
  int tok = blockIdx.x * 8 + wv;
  if (tok >= T) return;
  const float* xr = x + (size_t)tok * D;
  float* orow = out + (size_t)tok * D;
  const float* rrow = res ? res + (size_t)tok * D : nullptr;
  int ng = D / G;
  for (int g = 0; g < ng; g++) {
    float sm = 0.f, sq = 0.f;
    for (int e = g * G + lane; e < (g + 1) * G; e += 32) {
      float v = xr[e]; sm += v; sq += v * v;
    }
    for (int o = 16; o > 0; o >>= 1) {
      sm += __shfl_xor(sm, o, 32);
      sq += __shfl_xor(sq, o, 32);
    }
    float mu = sm / (float)G;
    float var = sq / (float)G - mu * mu;
    float rstd = rsqrtf(fmaxf(var, 0.f) + 1e-5f);
    for (int e = g * G + lane; e < (g + 1) * G; e += 32) {
      float o2 = (xr[e] - mu) * rstd * w[e];
      if (rrow) o2 += rrow[e];
      orow[e] = o2;
      if (outb) outb[(size_t)tok * D + e] = f2bf(o2);
    }
  }
}

// ---------------------------------------------------------------------------
// Depthwise causal conv (K=4) + SiLU; optional bf16 copy.
// ---------------------------------------------------------------------------
__global__ void k_conv_silu(const float* __restrict__ x, int ldx,
                            const float* __restrict__ w, const float* __restrict__ cb,
                            float* __restrict__ y, int ldy,
                            unsigned short* __restrict__ yb, int C, int total)
{
  int id = blockIdx.x * 256 + threadIdx.x;
  if (id >= total) return;
  int c = id % C, tok = id / C;
  int b = tok >> 10, s = tok & 1023;
  float acc = cb[c];
#pragma unroll
  for (int j = 0; j < 4; j++) {
    int ss = s - 3 + j;
    if (ss >= 0) acc += x[((size_t)(b * SEQ + ss)) * ldx + c] * w[c * 4 + j];
  }
  float o = acc / (1.f + __expf(-acc));
  y[(size_t)tok * ldy + c] = o;
  if (yb) yb[(size_t)tok * (size_t)ldy + c] = f2bf(o);
}

// ---------------------------------------------------------------------------
// Block-diagonal 4x4 headwise projection; optional bf16 copy.
// ---------------------------------------------------------------------------
__global__ void k_headwise(const float* __restrict__ x, int ldx,
                           const float* __restrict__ w,
                           float* __restrict__ y, unsigned short* __restrict__ yb,
                           int total)
{
  int id = blockIdx.x * 256 + threadIdx.x;
  if (id >= total) return;
  int tok = id >> 8, nb = id & 255;
  const float* xr = x + (size_t)tok * ldx + nb * 4;
  const float* wr = w + nb * 16;
#pragma unroll
  for (int o = 0; o < 4; o++) {
    float s = 0.f;
#pragma unroll
    for (int i = 0; i < 4; i++) s += xr[i] * wr[o * 4 + i];
    size_t oi = (size_t)tok * DINNER + nb * 4 + o;
    y[oi] = s;
    if (yb) yb[oi] = f2bf(s);
  }
}

// ---------------------------------------------------------------------------
// ig/fg gate projections (3072 -> NH), output transposed to (bh, s)
// ---------------------------------------------------------------------------
__global__ void k_gates(const float* __restrict__ q, const float* __restrict__ k,
                        const float* __restrict__ v,
                        const float* __restrict__ igw, const float* __restrict__ igb,
                        const float* __restrict__ fgw, const float* __restrict__ fgb,
                        float* __restrict__ igT, float* __restrict__ fgT)
{
  int id = blockIdx.x * 256 + threadIdx.x;
  if (id >= NTOK * NHEAD) return;
  int tok = id >> 2, h = id & 3;
  const float* qr = q + (size_t)tok * DINNER;
  const float* kr = k + (size_t)tok * DINNER;
  const float* vr = v + (size_t)tok * DINNER;
  float si = igb[h], sf = fgb[h];
  for (int j = 0; j < DINNER; j++) { si += qr[j] * igw[j * 4 + h];          sf += qr[j] * fgw[j * 4 + h]; }
  for (int j = 0; j < DINNER; j++) { si += kr[j] * igw[(DINNER + j) * 4 + h]; sf += kr[j] * fgw[(DINNER + j) * 4 + h]; }
  for (int j = 0; j < DINNER; j++) { si += vr[j] * igw[(2 * DINNER + j) * 4 + h]; sf += vr[j] * fgw[(2 * DINNER + j) * 4 + h]; }
  int b = tok >> 10, s = tok & 1023;
  igT[(b * 4 + h) * SEQ + s] = si;
  fgT[(b * 4 + h) * SEQ + s] = sf;
}

// Sequential per-(b,h) scan: u[t], running max M[s], nf[s]=exp(-max_log_d[s])
__global__ void k_gatescan(const float* __restrict__ igT, const float* __restrict__ fgT,
                           float* __restrict__ u, float* __restrict__ Mr,
                           float* __restrict__ nf)
{
  int bh = threadIdx.x;
  if (bh >= 16) return;
  float lf = 0.f, Mrun = 0.f;
  for (int s = 0; s < SEQ; s++) {
    float f = fgT[bh * SEQ + s];
    float ls = fminf(f, 0.f) - log1pf(__expf(-fabsf(f)));   // log_sigmoid
    lf += ls;
    float us = igT[bh * SEQ + s] - lf;
    Mrun = (s == 0) ? us : fmaxf(Mrun, us);
    u[bh * SEQ + s] = us;
    Mr[bh * SEQ + s] = Mrun;
    nf[bh * SEQ + s] = __expf(-(lf + Mrun));
  }
}

// hs = (hn + skip*xc) * silu(z_gate) -> bf16 only (feeds down-proj GEMM)
__global__ void k_hs(const float* __restrict__ hn, const float* __restrict__ xc,
                     const float* __restrict__ skip, const float* __restrict__ up,
                     unsigned short* __restrict__ hb, int total)
{
  int id = blockIdx.x * 256 + threadIdx.x;
  if (id >= total) return;
  int row = id >> 10, c = id & 1023;
  float v = hn[id] + skip[c] * xc[id];
  float zg = up[(size_t)row * 2048 + 1024 + c];
  float sz = zg / (1.f + __expf(-zg));
  hb[id] = f2bf(v * sz);
}

// gelu(gate)*up -> bf16 (feeds ff_down GEMM)
__global__ void k_gelu_mul(const float* __restrict__ gu, unsigned short* __restrict__ fb,
                           int total)
{
  int id = blockIdx.x * 256 + threadIdx.x;
  if (id >= total) return;
  int row = id / FFDIM, c = id - row * FFDIM;
  float g = gu[(size_t)row * (2 * FFDIM) + c];
  float u2 = gu[(size_t)row * (2 * FFDIM) + FFDIM + c];
  float o = 0.5f * g * (1.f + erff(g * 0.70710678f)) * u2;
  fb[id] = f2bf(o);
}

__global__ void k_cvt(const float* __restrict__ s, unsigned short* __restrict__ d, int n)
{
  int id = blockIdx.x * 256 + threadIdx.x;
  if (id < n) d[id] = f2bf(s[id]);
}

// transpose + convert: src (R x C) f32 -> dst (C x R) bf16
__global__ void k_tcvt(const float* __restrict__ s, unsigned short* __restrict__ d,
                       int R, int C)
{
  int id = blockIdx.x * 256 + threadIdx.x;
  if (id >= R * C) return;
  int r = id / C, c = id - r * C;
  d[(size_t)c * R + r] = f2bf(s[(size_t)r * C + c]);
}

// v (b,s,NH*256) f32 -> vT per (b,h): (256 x 1024) bf16
__global__ void k_vT(const float* __restrict__ v, unsigned short* __restrict__ vT)
{
  int id = blockIdx.x * 256 + threadIdx.x;     // 16*256*1024
  if (id >= 16 * DHM * SEQ) return;
  int t = id & 1023;
  int rest = id >> 10;
  int d = rest & 255;
  int bh = rest >> 8;
  int b = bh >> 2, h = bh & 3;
  vT[id] = f2bf(v[((size_t)(b * SEQ + t)) * DINNER + h * DHM + d]);
}

// rk (NH,DHS,4,DHS) -> rkT[(h*512+g*128+k)*128 + d]
__global__ void k_rkprep(const float* __restrict__ rk, float* __restrict__ rkT)
{
  int id = blockIdx.x * 256 + threadIdx.x;
  if (id >= 4 * 512 * 128) return;
  int d = id & 127;
  int gk = (id >> 7) & 511;
  int h = id >> 16;
  int g = gk >> 7, kk = gk & 127;
  rkT[id] = rk[(((size_t)(h * 128 + d)) * 4 + g) * 128 + kk];
}

// ---------------------------------------------------------------------------
// sLSTM recurrence: one block per (b,h), 512 threads (g,k), h-state in LDS.
// ---------------------------------------------------------------------------
__global__ __launch_bounds__(512) void k_slstm(
    const float* __restrict__ wx,    // (tok, 2048): h*512 + g*128 + k
    const float* __restrict__ rkT,   // (h*512+gk, 128)
    const float* __restrict__ bias,  // (NH,4,DHS)
    float* __restrict__ y)           // (tok, 512): h*128 + k
{
  int b = blockIdx.x >> 2, h = blockIdx.x & 3;
  int tid = threadIdx.x;
  __shared__ __align__(16) float hbuf[128];
  __shared__ float raw[512];
  float c = 0.f, n = 0.f, m = 0.f;
  if (tid < 128) hbuf[tid] = 0.f;
  __syncthreads();
  const f32x4* r4 = (const f32x4*)(rkT + (size_t)(h * 512 + tid) * 128);
  float bval = bias[(h * 4 + (tid >> 7)) * 128 + (tid & 127)];
  for (int s = 0; s < SEQ; s++) {
    float accv = wx[((size_t)(b * SEQ + s)) * 2048 + h * 512 + tid] + bval;
    const f32x4* h4 = (const f32x4*)hbuf;
#pragma unroll 8
    for (int d = 0; d < 32; d++) {
      f32x4 hh = h4[d], rr = r4[d];
      accv += hh.x * rr.x + hh.y * rr.y + hh.z * rr.z + hh.w * rr.w;
    }
    raw[tid] = accv;
    __syncthreads();
    if (tid < 128) {
      float ir = raw[tid], fr = raw[128 + tid], zr = raw[256 + tid], orr = raw[384 + tid];
      float lsf = fminf(fr, 0.f) - log1pf(__expf(-fabsf(fr)));
      float lfm = m + lsf;
      float mn = (n == 0.f) ? ir : fmaxf(ir, lfm);
      float ii = __expf(ir - mn), ff = __expf(lfm - mn);
      c = ff * c + ii * tanhf(zr);
      n = ff * n + ii;
      float hn = (1.f / (1.f + __expf(-orr))) * c / n;
      hbuf[tid] = hn;
      y[((size_t)(b * SEQ + s)) * 512 + h * 128 + tid] = hn;
      m = mn;
    }
    __syncthreads();
  }
}

// ===========================================================================
// Host orchestration
// ===========================================================================
extern "C" void kernel_launch(void* const* d_in, const int* in_sizes, int n_in,
                              void* d_out, int out_size, void* d_ws, size_t ws_size,
                              hipStream_t stream)
{
  (void)in_sizes; (void)n_in; (void)out_size; (void)ws_size;
  auto in = [&](int i) { return (const float*)d_in[i]; };

  // ---- workspace arena ----------------------------------------------------
  char* base = (char*)d_ws;
  size_t off = 0;
  auto arena = [&](size_t bytes) -> void* {
    void* p = base + off;
    off = (off + bytes + 255) & ~(size_t)255;
    return p;
  };
  float* f_z   = (float*)arena((size_t)NTOK * 512 * 4);
  float* f_xin = (float*)arena((size_t)NTOK * 512 * 4);
  float* f_tmp = (float*)arena((size_t)NTOK * 2048 * 4);   // up / wx / ff_gu
  float* f_xc  = (float*)arena((size_t)NTOK * 1024 * 4);   // xc / y(slstm)
  float* f_q   = (float*)arena((size_t)NTOK * 1024 * 4);   // q, then h
  float* f_k   = (float*)arena((size_t)NTOK * 1024 * 4);
  float* f_v   = (float*)arena((size_t)NTOK * 1024 * 4);
  unsigned short* b_xin = (unsigned short*)arena((size_t)NTOK * 512 * 2);
  unsigned short* b_act = (unsigned short*)arena((size_t)NTOK * 1024 * 2); // x / hs
  unsigned short* b_q   = (unsigned short*)arena((size_t)NTOK * 1024 * 2); // also ff
  unsigned short* b_k   = (unsigned short*)arena((size_t)NTOK * 1024 * 2); // also xc(s)
  unsigned short* b_vT  = (unsigned short*)arena((size_t)NTOK * 1024 * 2);
  unsigned short* b_w1  = (unsigned short*)arena((size_t)2048 * 512 * 2);
  unsigned short* b_w2  = (unsigned short*)arena((size_t)1024 * 512 * 2);
  unsigned short* b_wsl = (unsigned short*)arena((size_t)4 * 65536 * 2);
  float* f_rkT = (float*)arena((size_t)262144 * 4);
  float* f_ig  = (float*)arena((size_t)16 * SEQ * 4);
  float* f_fg  = (float*)arena((size_t)16 * SEQ * 4);
  float* f_u   = (float*)arena((size_t)16 * SEQ * 4);
  float* f_M   = (float*)arena((size_t)16 * SEQ * 4);
  float* f_nf  = (float*)arena((size_t)16 * SEQ * 4);

  auto eg = [](int n) { return dim3((unsigned)((n + 255) / 256)); };
  auto gemm = [&](const unsigned short* A, int lda, const unsigned short* Bt, int ldb,
                  float* C, int ldc, int M, int N, int K,
                  const float* bias, const float* res, unsigned short* Cb) {
    k_gemm<<<dim3(N / 64, M / 64), dim3(128), 0, stream>>>(A, lda, Bt, ldb, C, ldc,
                                                           bias, res, Cb, K);
  };
  auto ln = [&](const float* x, const float* w, float* out, unsigned short* ob,
                const float* res, int D, int G) {
    k_ln<<<dim3(NTOK / 8), dim3(256), 0, stream>>>(x, w, out, ob, res, NTOK, D, G);
  };

  // ---- embed: z = x @ embed_w + embed_b ----------------------------------
  const float* x_in    = in(0);
  const float* embed_w = in(1);
  const float* embed_b = in(2);
  k_cvt<<<eg(NTOK * 256), dim3(256), 0, stream>>>(x_in, b_act, NTOK * 256);
  k_tcvt<<<eg(256 * 512), dim3(256), 0, stream>>>(embed_w, b_w1, 256, 512);
  gemm(b_act, 256, b_w1, 256, f_z, 512, NTOK, 512, 256, embed_b, nullptr, nullptr);

  int idx = 3;
  for (int blk = 0; blk < 4; blk++) {
    if (blk == 1) {
      // ================= sLSTM block =================
      const float* ln1_w  = in(idx + 0);
      const float* conv_w = in(idx + 1);
      const float* conv_b = in(idx + 2);
      const float* iw     = in(idx + 3);
      const float* fw     = in(idx + 4);
      const float* zw     = in(idx + 5);
      const float* ow     = in(idx + 6);
      const float* rk     = in(idx + 7);
      const float* bias   = in(idx + 8);
      const float* gn_w   = in(idx + 9);
      const float* ln2_w  = in(idx + 10);
      const float* ff_up  = in(idx + 11);
      const float* ff_dn  = in(idx + 12);
      idx += 13;

      ln(f_z, ln1_w, f_xin, b_xin, nullptr, 512, 512);
      unsigned short* b_xc = b_k;
      k_conv_silu<<<eg(NTOK * 512), dim3(256), 0, stream>>>(
          f_xin, 512, conv_w, conv_b, f_xc, 512, b_xc, 512, NTOK * 512);

      const float* gw[4] = {iw, fw, zw, ow};
      for (int g = 0; g < 4; g++)
        k_cvt<<<eg(65536), dim3(256), 0, stream>>>(gw[g], b_wsl + g * 65536, 65536);

      // wx[(tok, h*512 + g*128 + k)] via 16 WMMA GEMMs (M=4096, N=128, K=128)
      for (int h = 0; h < 4; h++)
        for (int g = 0; g < 4; g++) {
          const unsigned short* A = (g < 2 ? b_xc : b_xin) + h * 128;
          gemm(A, 512, b_wsl + g * 65536 + h * 16384, 128,
               f_tmp + h * 512 + g * 128, 2048, NTOK, 128, 128,
               nullptr, nullptr, nullptr);
        }

      k_rkprep<<<eg(262144), dim3(256), 0, stream>>>(rk, f_rkT);
      k_slstm<<<dim3(16), dim3(512), 0, stream>>>(f_tmp, f_rkT, bias, f_xc);

      // z = z + mh_ln(y) ; then xf = ln(z)
      ln(f_xc, gn_w, f_z, nullptr, f_z, 512, 128);
      ln(f_z, ln2_w, f_xin, b_xin, nullptr, 512, 512);

      // FF: gu = xf @ ff_up ; ff = gelu(g)*u ; z += ff @ ff_down
      k_tcvt<<<eg(512 * 1408), dim3(256), 0, stream>>>(ff_up, b_w1, 512, 1408);
      gemm(b_xin, 512, b_w1, 512, f_tmp, 1408, NTOK, 1408, 512,
           nullptr, nullptr, nullptr);
      k_gelu_mul<<<eg(NTOK * FFDIM), dim3(256), 0, stream>>>(f_tmp, b_q, NTOK * FFDIM);
      k_tcvt<<<eg(704 * 512), dim3(256), 0, stream>>>(ff_dn, b_w2, 704, 512);
      gemm(b_q, 704, b_w2, 704, f_z, 512, NTOK, 512, 704, nullptr, f_z, nullptr);
    } else {
      // ================= mLSTM block =================
      const float* ln_w   = in(idx + 0);
      const float* up_w   = in(idx + 1);
      const float* conv_w = in(idx + 2);
      const float* conv_b = in(idx + 3);
      const float* q_w    = in(idx + 4);
      const float* k_w    = in(idx + 5);
      const float* v_w    = in(idx + 6);
      const float* ig_w   = in(idx + 7);
      const float* ig_b   = in(idx + 8);
      const float* fg_w   = in(idx + 9);
      const float* fg_b   = in(idx + 10);
      const float* onw    = in(idx + 11);
      const float* skip   = in(idx + 12);
      const float* down_w = in(idx + 13);
      idx += 14;

      ln(f_z, ln_w, f_xin, b_xin, nullptr, 512, 512);
      k_tcvt<<<eg(512 * 2048), dim3(256), 0, stream>>>(up_w, b_w1, 512, 2048);
      gemm(b_xin, 512, b_w1, 512, f_tmp, 2048, NTOK, 2048, 512,
           nullptr, nullptr, nullptr);    // [xm | z_gate]

      k_conv_silu<<<eg(NTOK * 1024), dim3(256), 0, stream>>>(
          f_tmp, 2048, conv_w, conv_b, f_xc, 1024, nullptr, 1024, NTOK * 1024);

      k_headwise<<<eg(NTOK * 256), dim3(256), 0, stream>>>(f_xc, 1024, q_w, f_q, b_q, NTOK * 256);
      k_headwise<<<eg(NTOK * 256), dim3(256), 0, stream>>>(f_xc, 1024, k_w, f_k, b_k, NTOK * 256);
      k_headwise<<<eg(NTOK * 256), dim3(256), 0, stream>>>(f_tmp, 2048, v_w, f_v, nullptr, NTOK * 256);

      k_gates<<<eg(NTOK * 4), dim3(256), 0, stream>>>(f_q, f_k, f_v, ig_w, ig_b,
                                                      fg_w, fg_b, f_ig, f_fg);
      k_gatescan<<<dim3(1), dim3(32), 0, stream>>>(f_ig, f_fg, f_u, f_M, f_nf);
      k_vT<<<eg(16 * 256 * 1024), dim3(256), 0, stream>>>(f_v, b_vT);

      // fused mLSTM cell -> h (reuses f_q storage)
      k_attn<<<dim3(16, 16), dim3(128), 0, stream>>>(b_q, b_k, b_vT, f_u, f_M, f_nf, f_q);

      ln(f_q, onw, f_q, nullptr, nullptr, 1024, 256);   // hn in place
      k_hs<<<eg(NTOK * 1024), dim3(256), 0, stream>>>(f_q, f_xc, skip, f_tmp,
                                                      b_act, NTOK * 1024);
      k_tcvt<<<eg(1024 * 512), dim3(256), 0, stream>>>(down_w, b_w2, 1024, 512);
      gemm(b_act, 1024, b_w2, 1024, f_z, 512, NTOK, 512, 1024, nullptr, f_z, nullptr);
    }
  }

  // ---- final LN into d_out ------------------------------------------------
  const float* post_w = in(idx);
  ln(f_z, post_w, (float*)d_out, nullptr, nullptr, 512, 512);
}
